// DynamicSpectralModulation_58291296141383
// MI455X (gfx1250) — matrix operand order repaired and checked by reading, not compiled, not run on hardware
//
#include <hip/hip_runtime.h>
#include <math.h>

// ---------------------------------------------------------------------------
// DynamicSpectralModulation on gfx1250 (MI455X), wave32.
//
// Pipeline (all on `stream`):
//   1) ln_stats_kernel   : per-token mean / rstd over C=768            (ws)
//   2) energy_kernel     : LN -> per-channel 14x14 -> rfft2 (direct
//                          14-pt DFTs in LDS) -> energy[b,c]           (ws)
//   3) mlp_wmma_kernel   : gelu(energy @ w1 + b1) via
//                          v_wmma_f32_16x16x32_f16 (f32 accum)         (ws)
//   4) mask_kernel       : hdn @ w2 + b2 -> mu,sigma -> 14x8 mask      (ws)
//   5) filter_kernel     : recompute LN+rfft2, apply mask, irfft2,
//                          residual combine, cls passthrough -> out
// ---------------------------------------------------------------------------

#define NTOK    196
#define TOKS    197
#define C_DIM   768
#define HID     192
#define BATCH   256
#define NSPEC   112          // 14 * 8 rfft2 bins
#define CHUNK   32           // channels per workgroup chunk
#define NCHUNK  24           // 768 / 32
#define PITCH_T 201          // padded 196 (odd -> conflict-free)
#define PITCH_S 113          // padded 112 (odd -> conflict-free)

typedef __attribute__((ext_vector_type(16))) _Float16 v16h;
typedef __attribute__((ext_vector_type(8)))  float    v8f;

// ---------------------------------------------------------------- LN stats --
__global__ void ln_stats_kernel(const float* __restrict__ x,
                                float* __restrict__ mu_out,
                                float* __restrict__ rs_out) {
  int wave = threadIdx.x >> 5;
  int lane = threadIdx.x & 31;
  int g = blockIdx.x * 8 + wave;            // token index over B*196
  if (g >= BATCH * NTOK) return;
  int b = g / NTOK, t = g % NTOK;
  const float* row = x + (size_t)b * TOKS * C_DIM + (size_t)(1 + t) * C_DIM;
  float s = 0.f, q = 0.f;
  for (int j = lane; j < C_DIM; j += 32) { float v = row[j]; s += v; q += v * v; }
  for (int off = 16; off > 0; off >>= 1) {
    s += __shfl_down(s, off, 32);
    q += __shfl_down(q, off, 32);
  }
  if (lane == 0) {
    float m   = s * (1.f / C_DIM);
    float var = q * (1.f / C_DIM) - m * m;
    mu_out[g] = m;
    rs_out[g] = rsqrtf(var + 1e-5f);
  }
}

// ------------------------------------------------------------ energy pass --
__global__ void energy_kernel(const float* __restrict__ x,
                              const float* __restrict__ gamma,
                              const float* __restrict__ beta,
                              const float* __restrict__ mu_ws,
                              const float* __restrict__ rs_ws,
                              float* __restrict__ energy) {
  __shared__ float sMu[NTOK], sRs[NTOK];
  __shared__ float sCt[14], sSt[14];
  __shared__ float sXn[CHUNK][PITCH_T];
  __shared__ float sXm[CHUNK];
  __shared__ float sPart[CHUNK][8];
  __shared__ float sAre[CHUNK][PITCH_S], sAim[CHUNK][PITCH_S];

  int b = blockIdx.x, c0 = blockIdx.y * CHUNK;
  int tid = threadIdx.x;

  if (tid < NTOK) { sMu[tid] = mu_ws[b * NTOK + tid]; sRs[tid] = rs_ws[b * NTOK + tid]; }
  if (tid < 14) {
    float a = 6.283185307179586f * (float)tid / 14.f;
    sCt[tid] = cosf(a); sSt[tid] = sinf(a);
  }
  __syncthreads();

  // load + LayerNorm into transposed (channel-major) LDS tile
  const float* xb = x + (size_t)b * TOKS * C_DIM;
  for (int idx = tid; idx < NTOK * CHUNK; idx += 256) {
    int t = idx >> 5, c = idx & 31;
    float v = xb[(size_t)(1 + t) * C_DIM + c0 + c];
    sXn[c][t] = (v - sMu[t]) * sRs[t] * gamma[c0 + c] + beta[c0 + c];
  }
  __syncthreads();

  // per-channel spatial mean (deterministic 8-way tree)
  {
    int c = tid >> 3, sub = tid & 7;
    float p = 0.f;
    for (int t = sub; t < NTOK; t += 8) p += sXn[c][t];
    sPart[c][sub] = p;
  }
  __syncthreads();
  if (tid < CHUNK) {
    float s = 0.f;
    for (int j = 0; j < 8; ++j) s += sPart[tid][j];
    sXm[tid] = s * (1.f / 196.f);
  }
  __syncthreads();

  // row DFT (along x), forward e^{-i theta}
  for (int idx = tid; idx < CHUNK * NSPEC; idx += 256) {
    int c = idx / NSPEC, o = idx % NSPEC;
    int y = o >> 3, kx = o & 7;
    float re = 0.f, im = 0.f, xm = sXm[c];
    for (int xx = 0; xx < 14; ++xx) {
      float v = sXn[c][y * 14 + xx] - xm;
      int w = (kx * xx) % 14;
      re += v * sCt[w];
      im -= v * sSt[w];
    }
    sAre[c][o] = re; sAim[c][o] = im;
  }
  __syncthreads();

  // column DFT + log1p|spec| accumulation (deterministic per channel)
  {
    int c = tid >> 3, sub = tid & 7;
    float esum = 0.f;
    for (int o = sub; o < NSPEC; o += 8) {
      int ky = o >> 3, kx = o & 7;
      float re = 0.f, im = 0.f;
      for (int y = 0; y < 14; ++y) {
        int w = (ky * y) % 14;
        float ar = sAre[c][y * 8 + kx], ai = sAim[c][y * 8 + kx];
        float cw = sCt[w], sw = sSt[w];
        re += ar * cw + ai * sw;     // (ar + i ai)(cw - i sw)
        im += ai * cw - ar * sw;
      }
      float mag = sqrtf(re * re + im * im) * (1.f / 14.f);  // ortho fwd scale
      esum += log1pf(mag);
    }
    sPart[c][sub] = esum;
  }
  __syncthreads();
  if (tid < CHUNK) {
    float s = 0.f;
    for (int j = 0; j < 8; ++j) s += sPart[tid][j];
    energy[(size_t)b * C_DIM + c0 + tid] = s * (1.f / 112.f);
  }
}

// ------------------------------------------------- MLP layer 1 via WMMA ----
// energy (256x768) @ w1 (768x192), f16 inputs / f32 accum, gelu(exact).
// One wave per 16x16 output tile: 16 M-tiles x 12 N-tiles = 192 waves.
__global__ void mlp_wmma_kernel(const float* __restrict__ energy,
                                const float* __restrict__ w1,
                                const float* __restrict__ b1,
                                float* __restrict__ hdn) {
  int wave = threadIdx.x >> 5;
  int lane = threadIdx.x & 31;
  int tile = blockIdx.x * 4 + wave;     // 0..191
  int tileM = tile & 15;                // 16 tiles of M=256
  int tileN = tile >> 4;                // 12 tiles of N=192
  int half  = lane >> 4;
  int l16   = lane & 15;
  int rowA  = tileM * 16 + l16;
  int colB  = tileN * 16 + l16;

  v8f acc = {0.f, 0.f, 0.f, 0.f, 0.f, 0.f, 0.f, 0.f};
  for (int k0 = 0; k0 < C_DIM; k0 += 32) {
    v16h a, bb;
#pragma unroll
    for (int e = 0; e < 16; ++e) {
      // A 16x32, contiguous k order (same as B): row = lane%16, K = half*16 + e
      int ka = k0 + half * 16 + e;
      a[e] = (_Float16)energy[(size_t)rowA * C_DIM + ka];
      // B 32x16, contiguous k order (same as A): col = lane%16, K = half*16 + e
      int kb = k0 + half * 16 + e;
      bb[e] = (_Float16)w1[(size_t)kb * HID + colB];
    }
    acc = __builtin_amdgcn_wmma_f32_16x16x32_f16(
        /*neg_a=*/false, a, /*neg_b=*/false, bb,
        /*c_mod=*/(short)0, acc, /*reuse_a=*/false, /*reuse_b=*/false);
  }

  float bias = b1[colB];
#pragma unroll
  for (int r = 0; r < 8; ++r) {
    int m = tileM * 16 + r + 8 * half;  // C layout: VGPR r -> M = r + 8*half
    float h = acc[r] + bias;
    float g = 0.5f * h * (1.f + erff(h * 0.70710678118654752f));
    hdn[(size_t)m * HID + colB] = g;
  }
}

// ---------------------------------------------- layer 2 + gaussian mask ----
__global__ void mask_kernel(const float* __restrict__ hdn,
                            const float* __restrict__ w2,
                            const float* __restrict__ b2,
                            float* __restrict__ mask) {
  __shared__ float sO[2];
  int b = blockIdx.x, tid = threadIdx.x;
  int wv = tid >> 5, lane = tid & 31;
  if (wv < 2) {                         // two waves: one per output column
    float s = 0.f;
    for (int n = lane; n < HID; n += 32) s += hdn[(size_t)b * HID + n] * w2[n * 2 + wv];
    for (int off = 16; off > 0; off >>= 1) s += __shfl_down(s, off, 32);
    if (lane == 0) sO[wv] = s + b2[wv];
  }
  __syncthreads();
  if (tid < NSPEC) {
    float o0 = sO[0], o1 = sO[1];
    float mu = fmaxf(9.899494936611665f / (1.f + expf(-o0)), 1.0f);  // MU_MAX*sigmoid
    float sp = (o1 > 20.f) ? o1 : log1pf(expf(o1));                  // softplus
    float sg = fmaxf(sp, 0.1f);
    float denom = 2.f * mu * mu * sg + 1e-6f;
    int ky = tid >> 3, kx = tid & 7;
    float fy = (ky <= 6) ? (float)ky : (float)(ky - 14);
    float d = fy * fy + (float)(kx * kx);
    mask[(size_t)b * NSPEC + tid] = expf(-d / denom);
  }
}

// ------------------------------------------------------------ filter pass --
__global__ void filter_kernel(const float* __restrict__ x,
                              const float* __restrict__ gamma,
                              const float* __restrict__ beta,
                              const float* __restrict__ mu_ws,
                              const float* __restrict__ rs_ws,
                              const float* __restrict__ mask,
                              const float* __restrict__ rsc_p,
                              float* __restrict__ out) {
  __shared__ float sMu[NTOK], sRs[NTOK], sMask[NSPEC];
  __shared__ float sCt[14], sSt[14];
  __shared__ float sXs[CHUNK][PITCH_T], sXn[CHUNK][PITCH_T];
  __shared__ float sXm[CHUNK];
  __shared__ float sPart[CHUNK][8];
  __shared__ float sAre[CHUNK][PITCH_S], sAim[CHUNK][PITCH_S];
  __shared__ float sBre[CHUNK][PITCH_S], sBim[CHUNK][PITCH_S];

  int b = blockIdx.x, c0 = blockIdx.y * CHUNK;
  int tid = threadIdx.x;

  if (tid < NTOK) { sMu[tid] = mu_ws[b * NTOK + tid]; sRs[tid] = rs_ws[b * NTOK + tid]; }
  if (tid < NSPEC) sMask[tid] = mask[(size_t)b * NSPEC + tid];
  if (tid < 14) {
    float a = 6.283185307179586f * (float)tid / 14.f;
    sCt[tid] = cosf(a); sSt[tid] = sinf(a);
  }
  __syncthreads();

  const float* xb = x + (size_t)b * TOKS * C_DIM;
  for (int idx = tid; idx < NTOK * CHUNK; idx += 256) {
    int t = idx >> 5, c = idx & 31;
    float v = xb[(size_t)(1 + t) * C_DIM + c0 + c];
    sXs[c][t] = v;
    sXn[c][t] = (v - sMu[t]) * sRs[t] * gamma[c0 + c] + beta[c0 + c];
  }
  __syncthreads();

  {
    int c = tid >> 3, sub = tid & 7;
    float p = 0.f;
    for (int t = sub; t < NTOK; t += 8) p += sXn[c][t];
    sPart[c][sub] = p;
  }
  __syncthreads();
  if (tid < CHUNK) {
    float s = 0.f;
    for (int j = 0; j < 8; ++j) s += sPart[tid][j];
    sXm[tid] = s * (1.f / 196.f);
  }
  __syncthreads();

  // forward row DFT
  for (int idx = tid; idx < CHUNK * NSPEC; idx += 256) {
    int c = idx / NSPEC, o = idx % NSPEC;
    int y = o >> 3, kx = o & 7;
    float re = 0.f, im = 0.f, xm = sXm[c];
    for (int xx = 0; xx < 14; ++xx) {
      float v = sXn[c][y * 14 + xx] - xm;
      int w = (kx * xx) % 14;
      re += v * sCt[w];
      im -= v * sSt[w];
    }
    sAre[c][o] = re; sAim[c][o] = im;
  }
  __syncthreads();

  // forward column DFT, ortho scale, apply mask
  for (int idx = tid; idx < CHUNK * NSPEC; idx += 256) {
    int c = idx / NSPEC, o = idx % NSPEC;
    int ky = o >> 3, kx = o & 7;
    float re = 0.f, im = 0.f;
    for (int y = 0; y < 14; ++y) {
      int w = (ky * y) % 14;
      float ar = sAre[c][y * 8 + kx], ai = sAim[c][y * 8 + kx];
      float cw = sCt[w], sw = sSt[w];
      re += ar * cw + ai * sw;
      im += ai * cw - ar * sw;
    }
    float m = sMask[o] * (1.f / 14.f);
    sBre[c][o] = re * m; sBim[c][o] = im * m;
  }
  __syncthreads();

  // inverse column DFT (full 14-point, e^{+i theta}) -> reuse A buffers
  for (int idx = tid; idx < CHUNK * NSPEC; idx += 256) {
    int c = idx / NSPEC, o = idx % NSPEC;
    int y = o >> 3, kx = o & 7;
    float re = 0.f, im = 0.f;
    for (int ky = 0; ky < 14; ++ky) {
      int w = (ky * y) % 14;
      float br = sBre[c][ky * 8 + kx], bi = sBim[c][ky * 8 + kx];
      float cw = sCt[w], sw = sSt[w];
      re += br * cw - bi * sw;       // (br + i bi)(cw + i sw)
      im += br * sw + bi * cw;
    }
    sAre[c][o] = re; sAim[c][o] = im;
  }
  __syncthreads();

  // inverse row DFT (Hermitian), ortho scale, residual combine, write out
  float rs = rsc_p[0];
  for (int idx = tid; idx < CHUNK * NTOK; idx += 256) {
    int c = idx & 31, t = idx >> 5;    // consecutive lanes -> consecutive c
    int y = t / 14, xx = t % 14;
    const float* zr = sAre[c];
    const float* zi = sAim[c];
    float acc = zr[y * 8 + 0] + zr[y * 8 + 7] * sCt[(7 * xx) % 14];
#pragma unroll
    for (int k = 1; k <= 6; ++k) {
      int w = (k * xx) % 14;
      acc += 2.f * (zr[y * 8 + k] * sCt[w] - zi[y * 8 + k] * sSt[w]);
    }
    float filt = acc * (1.f / 14.f) + sXm[c];
    float dv = filt - sXn[c][t];
    out[(size_t)b * TOKS * C_DIM + (size_t)(1 + t) * C_DIM + c0 + c] = sXs[c][t] + rs * dv;
  }

  if (blockIdx.y == 0) {               // cls token passthrough
    for (int j = tid; j < C_DIM; j += 256)
      out[(size_t)b * TOKS * C_DIM + j] = x[(size_t)b * TOKS * C_DIM + j];
  }
}

// ---------------------------------------------------------------- launch ---
extern "C" void kernel_launch(void* const* d_in, const int* in_sizes, int n_in,
                              void* d_out, int out_size, void* d_ws, size_t ws_size,
                              hipStream_t stream) {
  (void)in_sizes; (void)n_in; (void)out_size; (void)ws_size;
  const float* x     = (const float*)d_in[0];
  const float* gamma = (const float*)d_in[1];
  const float* beta  = (const float*)d_in[2];
  const float* w1    = (const float*)d_in[3];
  const float* b1    = (const float*)d_in[4];
  const float* w2    = (const float*)d_in[5];
  const float* b2    = (const float*)d_in[6];
  const float* rsc   = (const float*)d_in[7];
  float* out = (float*)d_out;
  float* ws  = (float*)d_ws;

  float* mu     = ws;                        // 256*196
  float* rstd   = ws + 50176;                // 256*196
  float* energy = ws + 100352;               // 256*768
  float* hdn    = ws + 296960;               // 256*192
  float* mask   = ws + 346112;               // 256*112  (total ~1.5 MB)

  ln_stats_kernel<<<6272, 256, 0, stream>>>(x, mu, rstd);
  dim3 gSpatial(BATCH, NCHUNK);
  energy_kernel<<<gSpatial, 256, 0, stream>>>(x, gamma, beta, mu, rstd, energy);
  mlp_wmma_kernel<<<48, 128, 0, stream>>>(energy, w1, b1, hdn);
  mask_kernel<<<BATCH, 128, 0, stream>>>(hdn, w2, b2, mask);
  filter_kernel<<<gSpatial, 256, 0, stream>>>(x, gamma, beta, mu, rstd, mask, rsc, out);
}
